// GNNEncoder_33294586479084
// MI455X (gfx1250) — matrix operand order repaired and checked by reading, not compiled, not run on hardware
//
#include <hip/hip_runtime.h>
#include <hip/hip_bf16.h>
#include <math.h>

// ---------------------------------------------------------------------------
// GNN encoder for MI455X (gfx1250, wave32).  All dense GEMM FLOPs go through
// V_WMMA_F32_16X16X32_F16 (f16 A/B, f32 accumulate).  Working set (~100MB)
// fits in the 192MB L2, so kernels are structured to maximize WMMA issue
// density: M-blocked 2x per wave so every B fragment (global_load_b128 from
// f16 weights) feeds two WMMAs; A tiles staged in LDS (ds_load_b128).
// ---------------------------------------------------------------------------

typedef __attribute__((ext_vector_type(16))) _Float16 v16h;
typedef __attribute__((ext_vector_type(8)))  _Float16 v8h;
typedef __attribute__((ext_vector_type(8)))  float    v8f;

constexpr int Bc  = 4;
constexpr int Nn  = 8192;
constexpr int Hh  = 256;
constexpr int Ee  = 8192;
constexpr int Mm  = 2048;
constexpr int ETt = 4;
constexpr int Tt  = 8;

constexpr int LDA = 264;   // LDS row stride (f16 elements), 16B-aligned rows, bank-skewed

// ---- WMMA helpers ----------------------------------------------------------

__device__ __forceinline__ v8f wmma16(v16h a, v16h b, v8f c) {
  // (neg_a, A, neg_b, B, c_mod, C, reuse_a, reuse_b)
  return __builtin_amdgcn_wmma_f32_16x16x32_f16(false, a, false, b, (short)0, c,
                                                false, false);
}

// A fragment (16x32 f16, MxK) from an LDS tile (16 rows x 256 cols, stride LDA).
// Layout: lanes 0-15 -> M=lane, K-base 0; lanes 16-31 -> M=lane-16, K-base 16;
// vector elems 0..15 = K kb..kb+15 (contiguous, matches load_b_frag).
__device__ __forceinline__ v16h load_a_frag(const _Float16* tile, int kt) {
  int lane = threadIdx.x & 31;
  int m    = lane & 15;
  int kb   = (lane & 16) ? 16 : 0;
  const _Float16* row = tile + m * LDA + kt * 32 + kb;
  v8h lo = *(const v8h*)(row);        // ds_load_b128
  v8h hi = *(const v8h*)(row + 8);    // ds_load_b128
  return __builtin_shufflevector(lo, hi, 0,1,2,3,4,5,6,7,8,9,10,11,12,13,14,15);
}

// B fragment (32x16 f16, KxN) computing X @ W^T:  Bfrag[k][n] = W[c0+n][kt*32+k].
// Lane n = lane&15 owns weight row c0+n; lane half selects K 0-15 / 16-31.
// 16 contiguous f16 = 32B contiguous -> 2x global_load_b128.
__device__ __forceinline__ v16h load_b_frag(const _Float16* __restrict__ W,
                                            int c0, int kt) {
  int lane  = threadIdx.x & 31;
  int n     = lane & 15;
  int khalf = (lane & 16);            // 0 or 16 elements
  return *(const v16h*)(W + (size_t)(c0 + n) * Hh + kt * 32 + khalf);
}

__device__ __forceinline__ float sigmoidf_(float x) {
  return 1.0f / (1.0f + __expf(-x));
}

// ---- small utility kernels -------------------------------------------------

__global__ void fill_f32(float* p, float v, size_t n) {
  size_t i = (size_t)blockIdx.x * blockDim.x + threadIdx.x;
  size_t stride = (size_t)gridDim.x * blockDim.x;
  for (; i < n; i += stride) p[i] = v;
}

__global__ void copy_f32(const float* __restrict__ s, float* __restrict__ d, size_t n4) {
  size_t i = (size_t)blockIdx.x * blockDim.x + threadIdx.x;
  size_t stride = (size_t)gridDim.x * blockDim.x;
  const float4* s4 = (const float4*)s;
  float4* d4 = (float4*)d;
  for (; i < n4; i += stride) d4[i] = s4[i];
}

__global__ void cvt_f32_to_f16(const float* __restrict__ s, _Float16* __restrict__ d,
                               size_t n) {
  size_t i = (size_t)blockIdx.x * blockDim.x + threadIdx.x;
  size_t stride = (size_t)gridDim.x * blockDim.x;
  for (; i < n; i += stride) d[i] = (_Float16)s[i];
}

// ---- message passing:  inc[tgt] += h[src] @ W_msg[e]^T + b_msg[e] ----------
// grid: (E/32, ET, B), block: 128 (4 waves).  Block = 32 edges x full H.
// Each wave: 2 M-tiles x 4 N-tiles = 8 accumulators; B frags reused 2x.

__global__ __launch_bounds__(128) void msg_kernel(
    const float* __restrict__ h,
    const int* __restrict__ e0, const int* __restrict__ e1,
    const int* __restrict__ e2, const int* __restrict__ e3,
    const _Float16* __restrict__ Wmsg,   // ET*H*H f16
    const float* __restrict__ bmsg,      // ET*H
    float* __restrict__ inc) {
  const int et = blockIdx.y;
  const int b  = blockIdx.z;
  const int ebase = blockIdx.x * 32;
  const int* edges = (et == 0) ? e0 : (et == 1) ? e1 : (et == 2) ? e2 : e3;

  __shared__ int s_src[32];
  __shared__ int s_tgt[32];
  __shared__ _Float16 sX[32 * LDA];

  const int tid = threadIdx.x;
  if (tid < 32) {
    size_t ei = ((size_t)b * Ee + ebase + tid) * 2;
    s_src[tid] = edges[ei + 0];
    s_tgt[tid] = edges[ei + 1];
  }
  __syncthreads();

  // stage gathered 32x256 tile of h, f32 -> f16 in LDS (4 threads per row)
  {
    int r    = tid >> 2;          // 0..31
    int cseg = (tid & 3) * 64;    // 0..192
    const float4* src4 = (const float4*)(h + ((size_t)b * Nn + s_src[r]) * Hh + cseg);
    _Float16* dst = &sX[r * LDA + cseg];
#pragma unroll
    for (int i = 0; i < 16; ++i) {
      float4 v = src4[i];
      dst[4 * i + 0] = (_Float16)v.x;
      dst[4 * i + 1] = (_Float16)v.y;
      dst[4 * i + 2] = (_Float16)v.z;
      dst[4 * i + 3] = (_Float16)v.w;
    }
  }
  __syncthreads();

  const int wave = tid >> 5;
  const int lane = tid & 31;
  const _Float16* Wt = Wmsg + (size_t)et * Hh * Hh;
  const int c0 = wave * 64;

  v8f acc[2][4];
#pragma unroll
  for (int mt = 0; mt < 2; ++mt)
#pragma unroll
    for (int nt = 0; nt < 4; ++nt) acc[mt][nt] = (v8f){};

#pragma unroll
  for (int kt = 0; kt < 8; ++kt) {
    v16h a0 = load_a_frag(sX, kt);
    v16h a1 = load_a_frag(sX + 16 * LDA, kt);
#pragma unroll
    for (int nt = 0; nt < 4; ++nt) {
      v16h bf = load_b_frag(Wt, c0 + nt * 16, kt);
      acc[0][nt] = wmma16(a0, bf, acc[0][nt]);
      acc[1][nt] = wmma16(a1, bf, acc[1][nt]);
    }
  }

  // C layout: VGPR r, lane l -> row m = r + 8*(l/16), col n = l%16; scatter-add.
  const int n     = lane & 15;
  const int mhalf = (lane & 16) ? 8 : 0;
#pragma unroll
  for (int nt = 0; nt < 4; ++nt) {
    int c = c0 + nt * 16 + n;
    float bv = bmsg[et * Hh + c];
#pragma unroll
    for (int mt = 0; mt < 2; ++mt) {
#pragma unroll
      for (int r = 0; r < 8; ++r) {
        int m = mt * 16 + r + mhalf;
        atomicAdd(&inc[((size_t)b * Nn + s_tgt[m]) * Hh + c], acc[mt][nt][r] + bv);
      }
    }
  }
}

// ---- GRU cell --------------------------------------------------------------
// grid: (N/32, 1, B), block: 128 (4 waves).  Block = 32 nodes x full H.
// Per wave and per 16-col tile: 2 M-tiles x 6 gate GEMMs = 12 accumulators;
// each of the 6 B fragments per k-tile feeds 2 WMMAs.

__global__ __launch_bounds__(128) void gru_kernel(
    const float* __restrict__ hin,
    const float* __restrict__ inc,
    const _Float16* __restrict__ Wih,   // 3H x H f16
    const _Float16* __restrict__ Whh,   // 3H x H f16
    const float* __restrict__ bih,      // 3H
    const float* __restrict__ bhh,      // 3H
    float* __restrict__ hout) {
  const int b = blockIdx.z;
  const int rbase = blockIdx.x * 32;

  __shared__ _Float16 sI[32 * LDA];
  __shared__ _Float16 sH[32 * LDA];

  const int tid = threadIdx.x;
  {
    int r    = tid >> 2;          // 0..31
    int cseg = (tid & 3) * 64;    // 0..192
    size_t base = ((size_t)b * Nn + rbase + r) * Hh + cseg;
    const float4* pi4 = (const float4*)(inc + base);
    const float4* ph4 = (const float4*)(hin + base);
    _Float16* di = &sI[r * LDA + cseg];
    _Float16* dh = &sH[r * LDA + cseg];
#pragma unroll
    for (int i = 0; i < 16; ++i) {
      float4 iv = pi4[i];
      float4 hv = ph4[i];
      di[4*i+0]=(_Float16)iv.x; di[4*i+1]=(_Float16)iv.y;
      di[4*i+2]=(_Float16)iv.z; di[4*i+3]=(_Float16)iv.w;
      dh[4*i+0]=(_Float16)hv.x; dh[4*i+1]=(_Float16)hv.y;
      dh[4*i+2]=(_Float16)hv.z; dh[4*i+3]=(_Float16)hv.w;
    }
  }
  __syncthreads();

  const int wave  = tid >> 5;
  const int lane  = tid & 31;
  const int n     = lane & 15;
  const int mhalf = (lane & 16) ? 8 : 0;

#pragma unroll 1
  for (int ct = 0; ct < 4; ++ct) {
    const int c0 = wave * 64 + ct * 16;   // column tile within gate dim H
    v8f acc[2][6];
#pragma unroll
    for (int mt = 0; mt < 2; ++mt)
#pragma unroll
      for (int g = 0; g < 6; ++g) acc[mt][g] = (v8f){};

#pragma unroll
    for (int kt = 0; kt < 8; ++kt) {
      v16h aI0 = load_a_frag(sI, kt);
      v16h aI1 = load_a_frag(sI + 16 * LDA, kt);
      v16h aH0 = load_a_frag(sH, kt);
      v16h aH1 = load_a_frag(sH + 16 * LDA, kt);
#pragma unroll
      for (int g = 0; g < 3; ++g) {
        v16h bi = load_b_frag(Wih, g * Hh + c0, kt);
        acc[0][g] = wmma16(aI0, bi, acc[0][g]);
        acc[1][g] = wmma16(aI1, bi, acc[1][g]);
        v16h bh = load_b_frag(Whh, g * Hh + c0, kt);
        acc[0][3 + g] = wmma16(aH0, bh, acc[0][3 + g]);
        acc[1][3 + g] = wmma16(aH1, bh, acc[1][3 + g]);
      }
    }

    const int c = c0 + n;
    float bir = bih[c], biz = bih[Hh + c], bin = bih[2 * Hh + c];
    float bhr = bhh[c], bhz = bhh[Hh + c], bhn = bhh[2 * Hh + c];
#pragma unroll
    for (int mt = 0; mt < 2; ++mt) {
#pragma unroll
      for (int r = 0; r < 8; ++r) {
        int m = mt * 16 + r + mhalf;
        size_t gidx = ((size_t)b * Nn + rbase + m) * Hh + c;
        float rg = sigmoidf_(acc[mt][0][r] + bir + acc[mt][3][r] + bhr);
        float zg = sigmoidf_(acc[mt][1][r] + biz + acc[mt][4][r] + bhz);
        float ng = tanhf(acc[mt][2][r] + bin + rg * (acc[mt][5][r] + bhn));
        float hv = hin[gidx];          // f32 h for the blend (L2-resident)
        hout[gidx] = (1.0f - zg) * ng + zg * hv;
      }
    }
  }
}

// ---- output: sel gather + mean-reduce for graph vector ---------------------

__global__ void out_gather(const float* __restrict__ h, const int* __restrict__ idx,
                           float* __restrict__ sel, float* __restrict__ gsum) {
  const int b = blockIdx.z;
  const int m = blockIdx.x;
  const int t = threadIdx.x;     // 64 threads * 4 cols
  const int node = idx[(size_t)b * Mm + m];
  const float4 v = ((const float4*)(h + ((size_t)b * Nn + node) * Hh))[t];
  ((float4*)(sel + ((size_t)b * Mm + m) * Hh))[t] = v;
  const float s = 1.0f / (float)Mm;
  atomicAdd(&gsum[b * Hh + 4 * t + 0], v.x * s);
  atomicAdd(&gsum[b * Hh + 4 * t + 1], v.y * s);
  atomicAdd(&gsum[b * Hh + 4 * t + 2], v.z * s);
  atomicAdd(&gsum[b * Hh + 4 * t + 3], v.w * s);
}

__global__ void out_finalize(const float* __restrict__ gsum, float* __restrict__ gvec) {
  int i = blockIdx.x * blockDim.x + threadIdx.x;
  if (i < Bc * Hh) gvec[i] = tanhf(gsum[i]);
}

// ---- driver ----------------------------------------------------------------

extern "C" void kernel_launch(void* const* d_in, const int* in_sizes, int n_in,
                              void* d_out, int out_size, void* d_ws, size_t ws_size,
                              hipStream_t stream) {
  const float* node_emb = (const float*)d_in[0];
  const int*   out_idx  = (const int*)d_in[2];
  const int*   e0 = (const int*)d_in[3];
  const int*   e1 = (const int*)d_in[4];
  const int*   e2 = (const int*)d_in[5];
  const int*   e3 = (const int*)d_in[6];
  const float* W_msg = (const float*)d_in[7];
  const float* b_msg = (const float*)d_in[8];
  const float* W_ih  = (const float*)d_in[9];
  const float* W_hh  = (const float*)d_in[10];
  const float* b_ih  = (const float*)d_in[11];
  const float* b_hh  = (const float*)d_in[12];

  const size_t nh = (size_t)Bc * Nn * Hh;

  float* hA   = (float*)d_ws;
  float* hB   = hA + nh;
  float* inc  = hB + nh;
  float* gsum = inc + nh;
  _Float16* Wmsg_h = (_Float16*)(gsum + Bc * Hh);
  _Float16* Wih_h  = Wmsg_h + (size_t)ETt * Hh * Hh;
  _Float16* Whh_h  = Wih_h + (size_t)3 * Hh * Hh;

  // weights -> f16 (tiny, once per launch)
  cvt_f32_to_f16<<<256, 256, 0, stream>>>(W_msg, Wmsg_h, (size_t)ETt * Hh * Hh);
  cvt_f32_to_f16<<<256, 256, 0, stream>>>(W_ih,  Wih_h,  (size_t)3 * Hh * Hh);
  cvt_f32_to_f16<<<256, 256, 0, stream>>>(W_hh,  Whh_h,  (size_t)3 * Hh * Hh);

  // h0 = node_embedding
  copy_f32<<<1024, 256, 0, stream>>>(node_emb, hA, nh / 4);

  float* cur = hA;
  float* nxt = hB;
  for (int t = 0; t < Tt; ++t) {
    fill_f32<<<1024, 256, 0, stream>>>(inc, 0.0f, nh);
    msg_kernel<<<dim3(Ee / 32, ETt, Bc), 128, 0, stream>>>(
        cur, e0, e1, e2, e3, Wmsg_h, b_msg, inc);
    gru_kernel<<<dim3(Nn / 32, 1, Bc), 128, 0, stream>>>(
        cur, inc, Wih_h, Whh_h, b_ih, b_hh, nxt);
    float* tmp = cur; cur = nxt; nxt = tmp;
  }

  // outputs: sel (B*M*H) | node_mask (B*M) | graph_vec (B*H), all f32
  float* sel  = (float*)d_out;
  float* mask = sel + (size_t)Bc * Mm * Hh;
  float* gvec = mask + (size_t)Bc * Mm;

  fill_f32<<<4, 256, 0, stream>>>(gsum, 0.0f, (size_t)Bc * Hh);
  out_gather<<<dim3(Mm, 1, Bc), 64, 0, stream>>>(cur, out_idx, sel, gsum);
  out_finalize<<<(Bc * Hh + 255) / 256, 256, 0, stream>>>(gsum, gvec);
  fill_f32<<<32, 256, 0, stream>>>(mask, 1.0f, (size_t)Bc * Mm);
}
